// BilinearLayer_74732430950505
// MI455X (gfx1250) — matrix operand order repaired and checked by reading, not compiled, not run on hardware
//
#include <hip/hip_runtime.h>

// ---------------------------------------------------------------------------
// Types for CDNA5 WMMA (wave32)
// ---------------------------------------------------------------------------
typedef __attribute__((ext_vector_type(16))) __bf16 v16bf;
typedef __attribute__((ext_vector_type(8)))  float  v8f;
typedef __attribute__((ext_vector_type(8)))  unsigned short us8;
typedef __attribute__((ext_vector_type(4)))  unsigned int ui4;
typedef __attribute__((ext_vector_type(8)))  int i8v;
typedef __attribute__((ext_vector_type(4)))  int i4v;

union FragBF {
    v16bf v;
    us8   h[2];
};

__device__ __forceinline__ unsigned short f2bf(float x) {
    union { float f; unsigned u; } v;
    v.f = x;
    unsigned r = v.u + 0x7FFFu + ((v.u >> 16) & 1u);
    return (unsigned short)(r >> 16);
}

// Problem constants
#define BB   64
#define LQ   128
#define LK   1024
#define EE   768
#define HH   6
#define HD   128
#define D2   64
#define NTIL 16          // LK / 64 row tiles in attention GEMM
#define BHN  (BB * HH)   // 384

// ---------------------------------------------------------------------------
// Weight transpose + f32 -> bf16 : Wt[n*K + k] = bf16(W[k*N + n])
// ---------------------------------------------------------------------------
__global__ void transpose_bf16_k(const float* __restrict__ W,
                                 unsigned short* __restrict__ Wt,
                                 int K, int N) {
    long tot = (long)K * N;
    for (long i = (long)blockIdx.x * blockDim.x + threadIdx.x; i < tot;
         i += (long)gridDim.x * blockDim.x) {
        long n = i / K;
        long k = i - n * K;
        Wt[i] = f2bf(W[k * N + n]);
    }
}

// f32 -> bf16 elementwise
__global__ void convert_bf16_k(const float* __restrict__ X,
                               unsigned short* __restrict__ Y, long n) {
    for (long i = (long)blockIdx.x * blockDim.x + threadIdx.x; i < n;
         i += (long)gridDim.x * blockDim.x)
        Y[i] = f2bf(X[i]);
}

// ---------------------------------------------------------------------------
// q = masked mean over LQ ; writes f32 + bf16
// ---------------------------------------------------------------------------
__global__ __launch_bounds__(256) void qmean_k(const float* __restrict__ q_feat,
                                               const float* __restrict__ qmask,
                                               float* __restrict__ qf,
                                               unsigned short* __restrict__ qb) {
    __shared__ float smr[LQ];
    __shared__ float smv[LQ];
    int b = blockIdx.x, tid = threadIdx.x;
    if (tid < LQ) { smv[tid] = qmask[b * LQ + tid]; smr[tid] = smv[tid]; }
    __syncthreads();
    for (int s = LQ / 2; s > 0; s >>= 1) {
        if (tid < s) smr[tid] += smr[tid + s];
        __syncthreads();
    }
    float inv = 1.0f / smr[0];
    for (int c = tid; c < EE; c += 256) {
        float s = 0.0f;
        for (int l = 0; l < LQ; ++l)
            s += q_feat[((long)b * LQ + l) * EE + c] * smv[l];
        float v = s * inv;
        qf[b * EE + c] = v;
        qb[b * EE + c] = f2bf(v);
    }
}

// k_mask row sums
__global__ __launch_bounds__(256) void msum_k(const float* __restrict__ kmask,
                                              float* __restrict__ msum) {
    __shared__ float sd[256];
    int b = blockIdx.x, tid = threadIdx.x;
    float s = 0.0f;
    for (int i = tid; i < LK; i += 256) s += kmask[b * LK + i];
    sd[tid] = s;
    __syncthreads();
    for (int st = 128; st > 0; st >>= 1) {
        if (tid < st) sd[tid] += sd[tid + st];
        __syncthreads();
    }
    if (tid == 0) msum[b] = sd[0];
}

// ---------------------------------------------------------------------------
// Main bf16 WMMA GEMM.  C(M x 768) = A(M x K) * Bt^T  (+ epilogue)
//   A is bf16 row-major, optionally split: k < K0 reads A0[(row>>a0_shift)*K0+k]
//   Bt is bf16 N-major [768][K]
//   epi: 0 = bias+tanh+groupnorm(128-col head) ; 1 = bias ; 2 = relu(+bias)+resid
// Block tile 64x128, K-step 32, 256 threads = 8 waves (2m x 4n), wave tile 32x32.
// ---------------------------------------------------------------------------
__global__ __launch_bounds__(256) void gemm_bf16_k(
    const unsigned short* __restrict__ A0, int K0, int a0_shift,
    const unsigned short* __restrict__ A1,
    const unsigned short* __restrict__ Bt,
    const float* __restrict__ bias,
    const float* __restrict__ gamma, const float* __restrict__ beta,
    const float* __restrict__ resid,
    float* __restrict__ Cf, unsigned short* __restrict__ Cbf,
    int K, int epi) {
    __shared__ __align__(16) unsigned short As[64][40];
    __shared__ __align__(16) unsigned short Bs[128][40];
    __shared__ float Csh[64][128];
    __shared__ float red[64][8];

    const int tid = threadIdx.x;
    const int lane = tid & 31;
    const int wid = tid >> 5;
    const int wm = wid & 1;
    const int wn = wid >> 1;
    const int hk = lane >> 4;
    const int l16 = lane & 15;
    const long row0 = (long)blockIdx.x * 64;
    const int col0 = blockIdx.y * 128;
    const int K1 = K - K0;

    const v8f vz = {0.f, 0.f, 0.f, 0.f, 0.f, 0.f, 0.f, 0.f};
    v8f acc[2][2];
    acc[0][0] = vz; acc[0][1] = vz; acc[1][0] = vz; acc[1][1] = vz;

    const int ar = tid >> 2;
    const int ak = (tid & 3) * 8;
    const int bn = tid >> 1;
    const int bk = (tid & 1) * 16;

    for (int kb = 0; kb < K; kb += 32) {
        {   // stage A tile 64x32 (one us8 per thread)
            long grow = row0 + ar;
            int gk = kb + ak;
            const unsigned short* src;
            if (gk < K0) src = A0 + (grow >> a0_shift) * (long)K0 + gk;
            else         src = A1 + grow * (long)K1 + (gk - K0);
            *(us8*)&As[ar][ak] = *(const us8*)src;
        }
        {   // stage B tile 128x32 (two us8 per thread)
            const unsigned short* src = Bt + (long)(col0 + bn) * K + kb + bk;
            *(us8*)&Bs[bn][bk]     = *(const us8*)src;
            *(us8*)&Bs[bn][bk + 8] = *(const us8*)(src + 8);
        }
        // gfx1250 prefetch of staging sources two K-steps ahead
        if (kb + 64 < K) {
            int pk = kb + 64 + ak;
            if (pk >= K0)
                __builtin_prefetch(A1 + (row0 + ar) * (long)K1 + (pk - K0), 0, 1);
            __builtin_prefetch(Bt + (long)(col0 + bn) * K + kb + 64 + bk, 0, 1);
        }
        __syncthreads();
        FragBF af[2], bf[2];
#pragma unroll
        for (int mi = 0; mi < 2; ++mi) {
            int r = wm * 32 + mi * 16 + l16;
            af[mi].h[0] = *(const us8*)&As[r][hk * 16];
            af[mi].h[1] = *(const us8*)&As[r][hk * 16 + 8];
        }
#pragma unroll
        for (int ni = 0; ni < 2; ++ni) {
            int n = wn * 32 + ni * 16 + l16;
            bf[ni].h[0] = *(const us8*)&Bs[n][hk * 16];
            bf[ni].h[1] = *(const us8*)&Bs[n][hk * 16 + 8];
        }
#pragma unroll
        for (int mi = 0; mi < 2; ++mi)
#pragma unroll
            for (int ni = 0; ni < 2; ++ni)
                acc[mi][ni] = __builtin_amdgcn_wmma_f32_16x16x32_bf16(
                    false, af[mi].v, false, bf[ni].v, (short)0, acc[mi][ni],
                    false, false);
        __syncthreads();
    }

    // write accumulators (+bias, +tanh) through LDS
#pragma unroll
    for (int mi = 0; mi < 2; ++mi)
#pragma unroll
        for (int ni = 0; ni < 2; ++ni) {
            int rb = wm * 32 + mi * 16 + hk * 8;
            int c = wn * 32 + ni * 16 + l16;
            float bv = bias[col0 + c];
#pragma unroll
            for (int r = 0; r < 8; ++r) {
                float v = acc[mi][ni][r] + bv;
                if (epi == 0) v = tanhf(v);
                Csh[rb + r][c] = v;
            }
        }
    __syncthreads();

    if (epi == 0) {  // per-row groupnorm over this 128-col head tile
        int row = tid >> 2, q = tid & 3;
        float s = 0.f, ss = 0.f;
        for (int c = q * 32; c < q * 32 + 32; ++c) {
            float v = Csh[row][c];
            s += v; ss += v * v;
        }
        red[row][q] = s; red[row][q + 4] = ss;
        __syncthreads();
        if (q == 0) {
            float S  = red[row][0] + red[row][1] + red[row][2] + red[row][3];
            float SS = red[row][4] + red[row][5] + red[row][6] + red[row][7];
            float mean = S * (1.0f / 128.0f);
            float var  = SS * (1.0f / 128.0f) - mean * mean;
            red[row][0] = mean;
            red[row][1] = rsqrtf(var + 1e-5f);
        }
        __syncthreads();
    }

    for (int i = tid; i < 64 * 128; i += 256) {
        int r = i >> 7, c = i & 127;
        float v = Csh[r][c];
        long o = (row0 + r) * EE + col0 + c;
        if (epi == 0)
            v = (v - red[r][0]) * red[r][1] * gamma[col0 + c] + beta[col0 + c];
        else if (epi == 2)
            v = fmaxf(v, 0.f) + resid[o];
        Cf[o] = v;
        if (Cbf) Cbf[o] = f2bf(v);
    }
}

// ---------------------------------------------------------------------------
// Attention GEMM per (b,h): basic = relu((qp .* kp) @ Wab + bab) for a 64-row
// tile; fused logits (= basic @ Wal + bal) and masked pool partial sums.
// Block 64 rows x 64 cols, K=128 in 4 steps; 8 waves (4m x 2n), wave 16x32.
// B tile (Wab, 64x128 bf16 with padded LDS rows) is loaded by the Tensor Data
// Mover: 2D tile, data_size=2B, pad 4 DWORDs every 64 DWORDs (row = 256B).
// ---------------------------------------------------------------------------
__global__ __launch_bounds__(256) void att_gemm_k(
    const float* __restrict__ qp, const float* __restrict__ kp,
    const unsigned short* __restrict__ WabT,   // [64][128] bf16 n-major
    const float* __restrict__ bab, const float* __restrict__ Wal,
    const float* __restrict__ bal, const float* __restrict__ kmask,
    float* __restrict__ logits,                // [BHN][LK]
    float* __restrict__ poolp) {               // [NTIL][BHN][64]
    __shared__ __align__(16) unsigned short As[64][40];
    __shared__ __align__(16) unsigned short Bs[64][136];
    __shared__ float Csh[64][64];
    __shared__ float qs[HD];
    __shared__ float wals[D2];

    const int tid = threadIdx.x;
    const int lane = tid & 31;
    const int wid = tid >> 5;
    const int wm = wid & 3;
    const int wn = wid >> 2;
    const int hk = lane >> 4;
    const int l16 = lane & 15;
    const int mtile = blockIdx.x;
    const int bh = blockIdx.y;
    const int b = bh / HH, hh = bh % HH;
    const int l0 = mtile * 64;

    if (tid < HD) qs[tid] = qp[b * EE + hh * HD + tid];
    if (tid >= 128 && tid < 128 + D2) wals[tid - 128] = Wal[tid - 128];

#if defined(__HIP_DEVICE_COMPILE__) && \
    __has_builtin(__builtin_amdgcn_tensor_load_to_lds)
    if (wid == 0) {
        // D# group 0: count=1, lds_addr, 57-bit global addr, type=2
        unsigned lds = (unsigned)(unsigned long long)(const void*)&Bs[0][0];
        unsigned long long ga = (unsigned long long)WabT;
        ui4 g0;
        g0[0] = 1u;
        g0[1] = lds;
        g0[2] = (unsigned)(ga & 0xFFFFFFFFu);
        g0[3] = (unsigned)((ga >> 32) & 0x01FFFFFFu) | (2u << 30);
        // D# group 1: data_size=2B, pad_enable, pad_interval=64 DW, pad=4 DW,
        // tensor_dim0=128, tensor_dim1=64, tile_dim0=128, tile_dim1=64,
        // tensor_dim0_stride=128
        i8v g1;
        g1[0] = (int)((1u << 16) | (1u << 20) | (5u << 22) | (3u << 25));
        g1[1] = (int)(128u << 16);   // tensor_dim0[15:0] << 16
        g1[2] = (int)(64u << 16);    // tensor_dim0 hi=0 | tensor_dim1[15:0]<<16
        g1[3] = (int)(128u << 16);   // tensor_dim1 hi=0 | tile_dim0 << 16
        g1[4] = 64;                  // tile_dim1
        g1[5] = 128;                 // tensor_dim0_stride lo
        g1[6] = 0;
        g1[7] = 0;
        i4v g2 = {0, 0, 0, 0};
        i4v g3 = {0, 0, 0, 0};
        i8v g4 = {0, 0, 0, 0, 0, 0, 0, 0};
        __builtin_amdgcn_tensor_load_to_lds(g0, g1, g2, g3, g4, 0);
        __builtin_amdgcn_s_wait_tensorcnt(0);
    }
#else
    {   // fallback: manual B tile load 64 x 128 bf16
        int n = tid >> 2;
        int kq = (tid & 3) * 32;
        const unsigned short* src = WabT + n * HD + kq;
        *(us8*)&Bs[n][kq]      = *(const us8*)src;
        *(us8*)&Bs[n][kq + 8]  = *(const us8*)(src + 8);
        *(us8*)&Bs[n][kq + 16] = *(const us8*)(src + 16);
        *(us8*)&Bs[n][kq + 24] = *(const us8*)(src + 24);
    }
#endif
    __syncthreads();

    const v8f vz = {0.f, 0.f, 0.f, 0.f, 0.f, 0.f, 0.f, 0.f};
    v8f acc[2];
    acc[0] = vz; acc[1] = vz;

    const int ar = tid >> 2;
    const int ak = (tid & 3) * 8;

    for (int kb = 0; kb < HD; kb += 32) {
        {   // stage A: (qp .* kp) -> bf16
            long row = (long)b * LK + l0 + ar;
            const float* src = kp + row * EE + hh * HD + kb + ak;
            float4 x0 = *(const float4*)src;
            float4 x1 = *(const float4*)(src + 4);
            int d = kb + ak;
            us8 t;
            t[0] = f2bf(x0.x * qs[d + 0]);
            t[1] = f2bf(x0.y * qs[d + 1]);
            t[2] = f2bf(x0.z * qs[d + 2]);
            t[3] = f2bf(x0.w * qs[d + 3]);
            t[4] = f2bf(x1.x * qs[d + 4]);
            t[5] = f2bf(x1.y * qs[d + 5]);
            t[6] = f2bf(x1.z * qs[d + 6]);
            t[7] = f2bf(x1.w * qs[d + 7]);
            *(us8*)&As[ar][ak] = t;
        }
        __syncthreads();
        FragBF af, bf0, bf1;
        int r = wm * 16 + l16;
        af.h[0] = *(const us8*)&As[r][hk * 16];
        af.h[1] = *(const us8*)&As[r][hk * 16 + 8];
        int n0 = wn * 32 + l16;
        bf0.h[0] = *(const us8*)&Bs[n0][kb + hk * 16];
        bf0.h[1] = *(const us8*)&Bs[n0][kb + hk * 16 + 8];
        int n1 = wn * 32 + 16 + l16;
        bf1.h[0] = *(const us8*)&Bs[n1][kb + hk * 16];
        bf1.h[1] = *(const us8*)&Bs[n1][kb + hk * 16 + 8];
        acc[0] = __builtin_amdgcn_wmma_f32_16x16x32_bf16(
            false, af.v, false, bf0.v, (short)0, acc[0], false, false);
        acc[1] = __builtin_amdgcn_wmma_f32_16x16x32_bf16(
            false, af.v, false, bf1.v, (short)0, acc[1], false, false);
        __syncthreads();
    }

#pragma unroll
    for (int ni = 0; ni < 2; ++ni) {
        int rb = wm * 16 + hk * 8;
        int c = wn * 32 + ni * 16 + l16;
        float bv = bab[c];
#pragma unroll
        for (int r = 0; r < 8; ++r)
            Csh[rb + r][c] = fmaxf(acc[ni][r] + bv, 0.f);
    }
    __syncthreads();

    if (tid < 64) {          // logits row dot Wal
        float s = bal[0];
        for (int c = 0; c < D2; ++c) s += Csh[tid][c] * wals[c];
        logits[(long)bh * LK + l0 + tid] = s;
    } else if (tid < 128) {  // masked pool partial over rows
        int c = tid - 64;
        float s = 0.f;
        for (int r = 0; r < 64; ++r)
            s += Csh[r][c] * kmask[b * LK + l0 + r];
        poolp[((long)mtile * BHN + bh) * D2 + c] = s;
    }
}

// deterministic pool reduce over row tiles
__global__ void pool_reduce_k(const float* __restrict__ poolp,
                              float* __restrict__ pool) {
    int bh = blockIdx.x, c = threadIdx.x;
    float s = 0.f;
    for (int t = 0; t < NTIL; ++t)
        s += poolp[((long)t * BHN + bh) * D2 + c];
    pool[bh * D2 + c] = s;
}

// masked softmax over LK, in place
__global__ __launch_bounds__(256) void softmax_k(float* __restrict__ logits,
                                                 const float* __restrict__ kmask) {
    __shared__ float sd[256];
    int bh = blockIdx.x, b = bh / HH, tid = threadIdx.x;
    float loc[4];
    float mx = -3.0e38f;
#pragma unroll
    for (int j = 0; j < 4; ++j) {
        int i = tid + j * 256;
        float x = logits[(long)bh * LK + i];
        if (kmask[b * LK + i] == 0.f) x = -1.0e9f;
        loc[j] = x;
        mx = fmaxf(mx, x);
    }
    sd[tid] = mx;
    __syncthreads();
    for (int s = 128; s > 0; s >>= 1) {
        if (tid < s) sd[tid] = fmaxf(sd[tid], sd[tid + s]);
        __syncthreads();
    }
    mx = sd[0];
    __syncthreads();
    float sum = 0.f;
#pragma unroll
    for (int j = 0; j < 4; ++j) {
        loc[j] = expf(loc[j] - mx);
        sum += loc[j];
    }
    sd[tid] = sum;
    __syncthreads();
    for (int s = 128; s > 0; s >>= 1) {
        if (tid < s) sd[tid] += sd[tid + s];
        __syncthreads();
    }
    float inv = 1.f / sd[0];
#pragma unroll
    for (int j = 0; j < 4; ++j)
        logits[(long)bh * LK + tid + j * 256] = loc[j] * inv;
}

// a_c = sigmoid(pool_mean @ Wac + bac)
__global__ __launch_bounds__(128) void ac_k(const float* __restrict__ pool,
                                            const float* __restrict__ msum,
                                            const float* __restrict__ Wac,
                                            const float* __restrict__ bac,
                                            float* __restrict__ ac) {
    __shared__ float pm[D2];
    int bh = blockIdx.x, b = bh / HH, tid = threadIdx.x;
    if (tid < D2) pm[tid] = pool[bh * D2 + tid] / msum[b];
    __syncthreads();
    float s = bac[tid];
    for (int j = 0; j < D2; ++j) s += pm[j] * Wac[j * HD + tid];
    ac[bh * HD + tid] = 1.f / (1.f + expf(-s));
}

// v2a = sum_l a_s * v2 ; out = v1 * v2a * a_c (writes f32 + bf16)
__global__ __launch_bounds__(128) void vout_k(const float* __restrict__ a_s,
                                              const float* __restrict__ V2,
                                              const float* __restrict__ v1,
                                              const float* __restrict__ ac,
                                              float* __restrict__ qf,
                                              unsigned short* __restrict__ qb) {
    int bh = blockIdx.x, b = bh / HH, hh = bh % HH, d = threadIdx.x;
    long base = (long)b * LK * EE + hh * HD + d;
    float acc = 0.f;
    for (int l = 0; l < LK; ++l)
        acc += a_s[(long)bh * LK + l] * V2[base + (long)l * EE];
    float val = v1[b * EE + hh * HD + d] * acc * ac[bh * HD + d];
    int o = b * EE + hh * HD + d;
    qf[o] = val;
    qb[o] = f2bf(val);
}

// row LayerNorm over 768 cols ; optional bf16 copy
__global__ __launch_bounds__(256) void ln768_k(const float* __restrict__ X,
                                               const float* __restrict__ g,
                                               const float* __restrict__ bta,
                                               float* __restrict__ Yf,
                                               unsigned short* __restrict__ Ybf) {
    __shared__ float s1[256], s2[256];
    long row = blockIdx.x;
    int tid = threadIdx.x;
    float v[3];
    float s = 0.f, ss = 0.f;
#pragma unroll
    for (int j = 0; j < 3; ++j) {
        float x = X[row * EE + tid + j * 256];
        v[j] = x;
        s += x;
        ss += x * x;
    }
    s1[tid] = s;
    s2[tid] = ss;
    __syncthreads();
    for (int st = 128; st > 0; st >>= 1) {
        if (tid < st) { s1[tid] += s1[tid + st]; s2[tid] += s2[tid + st]; }
        __syncthreads();
    }
    float mean = s1[0] * (1.0f / EE);
    float var  = s2[0] * (1.0f / EE) - mean * mean;
    float rstd = rsqrtf(var + 1e-5f);
#pragma unroll
    for (int j = 0; j < 3; ++j) {
        int c = tid + j * 256;
        float y = (v[j] - mean) * rstd * g[c] + bta[c];
        Yf[row * EE + c] = y;
        if (Ybf) Ybf[row * EE + c] = f2bf(y);
    }
}

// build bf16 concat [q0 | q1 | q2]  (64 x 2304)
__global__ void catbuild_k(const float* __restrict__ q0,
                           const float* __restrict__ q1,
                           const float* __restrict__ q2,
                           unsigned short* __restrict__ catb) {
    int i = blockIdx.x * blockDim.x + threadIdx.x;
    if (i >= BB * 3 * EE) return;
    int b = i / (3 * EE), c = i % (3 * EE);
    float v = (c < EE) ? q0[b * EE + c]
              : (c < 2 * EE) ? q1[b * EE + c - EE]
                             : q2[b * EE + c - 2 * EE];
    catb[i] = f2bf(v);
}

// ---------------------------------------------------------------------------
// Host orchestration
// ---------------------------------------------------------------------------
extern "C" void kernel_launch(void* const* d_in, const int* in_sizes, int n_in,
                              void* d_out, int out_size, void* d_ws, size_t ws_size,
                              hipStream_t stream) {
    (void)in_sizes; (void)n_in; (void)out_size; (void)ws_size;
    const float* q_feat = (const float*)d_in[0];
    const float* k_feats = (const float*)d_in[1];
    const float* q_mask = (const float*)d_in[2];
    const float* k_mask = (const float*)d_in[3];
    const float* Wq  = (const float*)d_in[4];
    const float* bq  = (const float*)d_in[5];
    const float* gq_g = (const float*)d_in[6];
    const float* gq_b = (const float*)d_in[7];
    const float* Wk  = (const float*)d_in[8];
    const float* bk  = (const float*)d_in[9];
    const float* gk_g = (const float*)d_in[10];
    const float* gk_b = (const float*)d_in[11];
    const float* Wv1 = (const float*)d_in[12];
    const float* bv1 = (const float*)d_in[13];
    const float* g1_g = (const float*)d_in[14];
    const float* g1_b = (const float*)d_in[15];
    const float* Wv2 = (const float*)d_in[16];
    const float* bv2 = (const float*)d_in[17];
    const float* g2_g = (const float*)d_in[18];
    const float* g2_b = (const float*)d_in[19];
    const float* Wab = (const float*)d_in[20];
    const float* bab = (const float*)d_in[21];
    const float* Wal = (const float*)d_in[22];
    const float* bal = (const float*)d_in[23];
    const float* Wac = (const float*)d_in[24];
    const float* bac = (const float*)d_in[25];
    const float* Wbi = (const float*)d_in[26];
    const float* bbi = (const float*)d_in[27];
    const float* lnb_g = (const float*)d_in[28];
    const float* lnb_b = (const float*)d_in[29];
    const float* Wp  = (const float*)d_in[30];
    const float* bp  = (const float*)d_in[31];
    const float* ln_g = (const float*)d_in[32];
    const float* ln_b = (const float*)d_in[33];

    size_t off = 0;
    auto alloc = [&](size_t bytes) -> void* {
        void* p = (char*)d_ws + off;
        off = (off + bytes + 255) & ~(size_t)255;
        return p;
    };
    unsigned short* WqT[2]; unsigned short* WkT[2];
    unsigned short* Wv1T[2]; unsigned short* Wv2T[2]; unsigned short* WabT[2];
    for (int l = 0; l < 2; ++l) {
        WqT[l]  = (unsigned short*)alloc((size_t)EE * EE * 2);
        WkT[l]  = (unsigned short*)alloc((size_t)EE * EE * 2);
        Wv1T[l] = (unsigned short*)alloc((size_t)EE * EE * 2);
        Wv2T[l] = (unsigned short*)alloc((size_t)EE * EE * 2);
        WabT[l] = (unsigned short*)alloc((size_t)D2 * HD * 2);
    }
    unsigned short* WbiT = (unsigned short*)alloc((size_t)EE * 2 * EE * 2);
    unsigned short* WpT  = (unsigned short*)alloc((size_t)EE * 3 * EE * 2);
    float* q0f = (float*)alloc((size_t)BB * EE * 4);
    float* q1f = (float*)alloc((size_t)BB * EE * 4);
    float* q2f = (float*)alloc((size_t)BB * EE * 4);
    unsigned short* q0b = (unsigned short*)alloc((size_t)BB * EE * 2);
    unsigned short* q1b = (unsigned short*)alloc((size_t)BB * EE * 2);
    unsigned short* q2b = (unsigned short*)alloc((size_t)BB * EE * 2);
    float* qpf = (float*)alloc((size_t)BB * EE * 4);
    float* v1f = (float*)alloc((size_t)BB * EE * 4);
    unsigned short* KBF = (unsigned short*)alloc((size_t)BB * LK * EE * 2);
    float* KF1 = (float*)alloc((size_t)BB * LK * EE * 4);
    float* KP  = (float*)alloc((size_t)BB * LK * EE * 4);   // also reused as T
    float* V2  = (float*)alloc((size_t)BB * LK * EE * 4);
    float* logits = (float*)alloc((size_t)BHN * LK * 4);
    float* poolp = (float*)alloc((size_t)NTIL * BHN * D2 * 4);
    float* pool  = (float*)alloc((size_t)BHN * D2 * 4);
    float* acbuf = (float*)alloc((size_t)BHN * HD * 4);
    float* msum = (float*)alloc((size_t)BB * 4);
    float* pproj = (float*)alloc((size_t)BB * EE * 4);
    unsigned short* catb = (unsigned short*)alloc((size_t)BB * 3 * EE * 2);

    // weight prep
    auto TKN = [&](const float* W, unsigned short* Wt, int K, int N) {
        long tot = (long)K * N;
        int g = (int)((tot + 255) / 256);
        if (g > 16384) g = 16384;
        transpose_bf16_k<<<g, 256, 0, stream>>>(W, Wt, K, N);
    };
    for (int l = 0; l < 2; ++l) {
        TKN(Wq  + (size_t)l * EE * EE, WqT[l],  EE, EE);
        TKN(Wk  + (size_t)l * EE * EE, WkT[l],  EE, EE);
        TKN(Wv1 + (size_t)l * EE * EE, Wv1T[l], EE, EE);
        TKN(Wv2 + (size_t)l * EE * EE, Wv2T[l], EE, EE);
        TKN(Wab + (size_t)l * HD * D2, WabT[l], HD, D2);
    }
    TKN(Wbi, WbiT, 2 * EE, EE);
    TKN(Wp,  WpT,  3 * EE, EE);

    convert_bf16_k<<<16384, 256, 0, stream>>>(k_feats, KBF, (long)BB * LK * EE);
    qmean_k<<<BB, 256, 0, stream>>>(q_feat, q_mask, q0f, q0b);
    msum_k<<<BB, 256, 0, stream>>>(k_mask, msum);

    auto GEMM = [&](const unsigned short* A0, int K0, int sh,
                    const unsigned short* A1, const unsigned short* Bt,
                    const float* bias, const float* gm, const float* bt,
                    const float* res, float* Cf, unsigned short* Cbf,
                    int M, int K, int epi) {
        dim3 grid(M / 64, EE / 128);
        gemm_bf16_k<<<grid, 256, 0, stream>>>(A0, K0, sh, A1, Bt, bias, gm, bt,
                                              res, Cf, Cbf, K, epi);
    };

    const float* kcur_f = k_feats;
    unsigned short* kcur_b = KBF;
    const unsigned short* qin_b = q0b;
    float* qlf[2] = {q1f, q2f};
    unsigned short* qlb[2] = {q1b, q2b};

    for (int l = 0; l < 2; ++l) {
        // q-side projections (M = 64) with fused tanh+groupnorm
        GEMM(nullptr, 0, 0, qin_b, WqT[l], bq + l * EE, gq_g + l * EE,
             gq_b + l * EE, nullptr, qpf, nullptr, BB, EE, 0);
        GEMM(nullptr, 0, 0, qin_b, Wv1T[l], bv1 + l * EE, g1_g + l * EE,
             g1_b + l * EE, nullptr, v1f, nullptr, BB, EE, 0);
        // k-side projections (M = 65536)
        GEMM(nullptr, 0, 0, kcur_b, WkT[l], bk + l * EE, gk_g + l * EE,
             gk_b + l * EE, nullptr, KP, nullptr, BB * LK, EE, 0);
        GEMM(nullptr, 0, 0, kcur_b, Wv2T[l], bv2 + l * EE, g2_g + l * EE,
             g2_b + l * EE, nullptr, V2, nullptr, BB * LK, EE, 0);
        // attention core
        att_gemm_k<<<dim3(NTIL, BHN), 256, 0, stream>>>(
            qpf, KP, WabT[l], bab + l * D2, Wal + l * D2, bal + l, k_mask,
            logits, poolp);
        pool_reduce_k<<<BHN, D2, 0, stream>>>(poolp, pool);
        softmax_k<<<BHN, 256, 0, stream>>>(logits, k_mask);
        ac_k<<<BHN, HD, 0, stream>>>(pool, msum, Wac + (size_t)l * D2 * HD,
                                     bac + l * HD, acbuf);
        vout_k<<<BHN, HD, 0, stream>>>(logits, V2, v1f, acbuf, qlf[l], qlb[l]);
        qin_b = qlb[l];
        if (l == 0) {  // k update: relu(cat@Wbi + bbi) + k, then LN
            GEMM(qlb[0], EE, 10, kcur_b, WbiT, bbi, nullptr, nullptr, kcur_f,
                 KP, nullptr, BB * LK, 2 * EE, 2);
            ln768_k<<<BB * LK, 256, 0, stream>>>(KP, lnb_g, lnb_b, KF1, KBF);
            kcur_f = KF1;
            kcur_b = KBF;
        }
    }

    // final projection + LN
    catbuild_k<<<(BB * 3 * EE + 255) / 256, 256, 0, stream>>>(q0f, q1f, q2f, catb);
    GEMM(nullptr, 0, 0, catb, WpT, bp, nullptr, nullptr, nullptr, pproj,
         nullptr, BB, 3 * EE, 1);
    ln768_k<<<BB, 256, 0, stream>>>(pproj, ln_g, ln_b, (float*)d_out, nullptr);
}
